// ModDeform2d_59751585022542
// MI455X (gfx1250) — matrix operand order repaired and checked by reading, not compiled, not run on hardware
//
#include <hip/hip_runtime.h>
#include <hip/hip_bf16.h>

// ---------------------------------------------------------------------------
// Modulated deformable conv (DCNv2) forward for MI455X (gfx1250, wave32).
//   B=4, C_IN=C_OUT=256, H=W=64, K=3, stride=1, pad=1, dil=1
// Main einsum = implicit GEMM M=16384, N=256, K=2304 -> 19.3 GFLOP; all data
// is L2-resident (192MB), so matrix-core bound: v_wmma_f32_16x16x32_f16 with
// f32 accumulation. B tiles are DMA'd into LDS by the Tensor Data Mover
// (tensor_load_to_lds + s_wait_tensorcnt), using D# padding to produce the
// bank-conflict-free [128][36] layout directly.
// ---------------------------------------------------------------------------

typedef _Float16 h16;
typedef __attribute__((ext_vector_type(16))) _Float16 v16h;
typedef __attribute__((ext_vector_type(2)))  _Float16 h2;
typedef __attribute__((ext_vector_type(8)))  float    v8f;
typedef unsigned int u32x4 __attribute__((ext_vector_type(4)));
typedef int          i32x8 __attribute__((ext_vector_type(8)));
typedef int          i32x4 __attribute__((ext_vector_type(4)));

static constexpr int kBatch = 4;
static constexpr int kCin   = 256;
static constexpr int kCout  = 256;
static constexpr int kH     = 64;
static constexpr int kW     = 64;
static constexpr int kHW    = kH * kW;          // 4096
static constexpr int kKK    = 9;

static constexpr int kModElems = kBatch * 27 * kHW;          // 442368 f32
static constexpr size_t kModBytes = (size_t)kModElems * 4;   // 1,769,472 B
static constexpr int kWtElems  = kKK * kCin * kCout;         // 589824 f16

// ---------------------------------------------------------------------------
// Kernel 0: mod = conv3x3(x, w_mod), pad=1. Raw 27 channels per pixel, f32.
// ---------------------------------------------------------------------------
__global__ __launch_bounds__(64) void mod_conv_kernel(
    const float* __restrict__ x, const float* __restrict__ wmod,
    float* __restrict__ modraw)
{
  const int bid = blockIdx.x;
  const int ch  = bid % 27;
  const int y   = (bid / 27) % kH;
  const int b   = bid / (27 * kH);
  const int xc  = threadIdx.x;

  const float* wp = wmod + ch * (kCin * kKK);
  float acc = 0.f;
  for (int c = 0; c < kCin; ++c) {
    const float* xp = x + ((b * kCin + c) << 12);
    const float* wc = wp + c * kKK;
#pragma unroll
    for (int ky = 0; ky < 3; ++ky) {
      const int iy = y + ky - 1;
      if (iy < 0 || iy >= kH) continue;
#pragma unroll
      for (int kx = 0; kx < 3; ++kx) {
        const int ix = xc + kx - 1;
        if (ix >= 0 && ix < kW)
          acc = fmaf(xp[(iy << 6) + ix], wc[ky * 3 + kx], acc);
      }
    }
  }
  modraw[((b * 27 + ch) << 12) + (y << 6) + xc] = acc;
}

// ---------------------------------------------------------------------------
// Kernel 1: repack w_def [O][C][3][3] f32 -> wt[kk][o][c] f16 (c contiguous),
// so the per-chunk B tile is a dense 2D tile TDM can fetch (rows = o).
// ---------------------------------------------------------------------------
__global__ __launch_bounds__(256) void wconv_kernel(
    const float* __restrict__ wdef, h16* __restrict__ wt)
{
  const int i = blockIdx.x * 256 + threadIdx.x;
  if (i >= kWtElems) return;
  const int c  = i & 255;
  const int o  = (i >> 8) & 255;
  const int kk = i >> 16;
  wt[i] = (h16)wdef[(o * kCin + c) * kKK + kk];
}

// ---------------------------------------------------------------------------
// Kernel 2: deformable implicit GEMM with WMMA + TDM weight staging.
//   Block: M-tile = 64 positions (one (b,y) row), N-tile = 128 outputs.
//   Loop: 9 taps x 8 chunks of 32 channels; 8 waves x 4 WMMA per chunk.
// ---------------------------------------------------------------------------
__global__ __launch_bounds__(256) void deform_gemm_kernel(
    const float* __restrict__ x, const float* __restrict__ modraw,
    const h16* __restrict__ wt, const float* __restrict__ bias,
    float* __restrict__ out)
{
  __shared__ __attribute__((aligned(16))) h16   Asub[64][40];    // [m][c]  (pad 32->40)
  __shared__ __attribute__((aligned(16))) h16   BsubT[128][36];  // [o][c]  (TDM pad 32->36)
  __shared__ __attribute__((aligned(16))) int   sIdx[4][64];     // clamped corner offsets
  __shared__ __attribute__((aligned(16))) float sW[4][64];       // mask-premult weights
  __shared__ __attribute__((aligned(16))) float Dtile[128][68];  // epilogue transpose

  const int t    = threadIdx.x;
  const int lane = t & 31;
  const int wave = t >> 5;             // 0..7 -> nt
  const int m0   = blockIdx.x << 6;    // 64 positions
  const int n0   = blockIdx.y << 7;    // 128 outputs
  const int b    = m0 >> 12;           // positions in a block share (b, y)
  const int yy   = (m0 >> 6) & 63;

  v8f acc[4] = {};                     // 4 M-subtiles x v8f

  const float* __restrict__ modb = modraw + (b * 27 << 12) + (yy << 6);
  const float* __restrict__ xb   = x + (b << 20);   // scalar base: 32-bit gathers

#if __has_builtin(__builtin_amdgcn_tensor_load_to_lds)
  // ---- TDM descriptor group1 (uniform for all chunks): 2-byte elems, tile
  //      32 x 128 rows, row stride 256 elems, LDS pad 2 DW per 16 DW row ----
  i32x8 g1;
  g1[0] = (1 << 16)      // data_size = 2B
        | (1 << 20)      // pad_enable
        | (3 << 22)      // pad_interval: 16 DWORDs (one 32-half row)
        | (1 << 25);     // pad_amount: 2 DWORDs  -> row stride 36 halfs
  g1[1] = 0;                       // atomic barrier addr / tensor_dim0 lo16 (=0)
  g1[2] = 0x10;                    // tensor_dim0 hi16 (1<<20, safely large)
  g1[3] = 0x10 | (32 << 16);       // tensor_dim1 hi16 (1<<20) | tile_dim0 = 32
  g1[4] = 128;                     // tile_dim1 = 128 rows (tile_dim2 = 0)
  g1[5] = 256;                     // tensor_dim0_stride = 256 elems
  g1[6] = 0;
  g1[7] = 0;
  const i32x4 gz4 = {0, 0, 0, 0};
  const i32x8 gz8 = {0, 0, 0, 0, 0, 0, 0, 0};
  const unsigned ldsB = (unsigned)(size_t)&BsubT[0][0];
  const unsigned long long wtBase = (unsigned long long)(size_t)wt;
#endif

  for (int kk = 0; kk < kKK; ++kk) {
    // --- per (position, tap) sampling parameters: 64 lanes, once per tap ---
    if (t < 64) {
      const int ky = kk / 3, kx = kk % 3;
      const float dy = modb[((2 * kk) << 12) + t];
      const float dx = modb[((2 * kk + 1) << 12) + t];
      const float mr = modb[((18 + kk) << 12) + t];
      const float msk = 1.f / (1.f + expf(-mr));
      const float py = (float)(yy - 1 + ky) + dy;
      const float px = (float)(t - 1 + kx) + dx;
      const float fy0 = floorf(py), fx0 = floorf(px);
      const int y0 = (int)fy0, x0 = (int)fx0;
      const float fy = py - fy0, fx = px - fx0;
      const int y0c = min(max(y0, 0), kH - 1);
      const int y1c = min(max(y0 + 1, 0), kH - 1);
      const int x0c = min(max(x0, 0), kW - 1);
      const int x1c = min(max(x0 + 1, 0), kW - 1);
      const float vy0 = (y0 >= 0 && y0 < kH) ? 1.f : 0.f;
      const float vy1 = (y0 + 1 >= 0 && y0 + 1 < kH) ? 1.f : 0.f;
      const float vx0 = (x0 >= 0 && x0 < kW) ? 1.f : 0.f;
      const float vx1 = (x0 + 1 >= 0 && x0 + 1 < kW) ? 1.f : 0.f;
      sIdx[0][t] = (y0c << 6) + x0c;  sW[0][t] = (1.f - fy) * (1.f - fx) * vy0 * vx0 * msk;
      sIdx[1][t] = (y0c << 6) + x1c;  sW[1][t] = (1.f - fy) * fx        * vy0 * vx1 * msk;
      sIdx[2][t] = (y1c << 6) + x0c;  sW[2][t] = fy * (1.f - fx)        * vy1 * vx0 * msk;
      sIdx[3][t] = (y1c << 6) + x1c;  sW[3][t] = fy * fx                * vy1 * vx1 * msk;
    }
    __syncthreads();

    // Params are invariant per thread for the whole tap: hoist to registers.
    const int   mloc = t & 63;
    const int   i0 = sIdx[0][mloc], i1 = sIdx[1][mloc];
    const int   i2 = sIdx[2][mloc], i3 = sIdx[3][mloc];
    const float w0 = sW[0][mloc], w1 = sW[1][mloc];
    const float w2 = sW[2][mloc], w3 = sW[3][mloc];

    for (int c0 = 0; c0 < kCin; c0 += 32) {
#if __has_builtin(__builtin_amdgcn_tensor_load_to_lds)
      // --- B tile via Tensor Data Mover (wave 0 issues one DMA) ---
      if (t < 32) {
        const unsigned long long gaddr =
            wtBase + 2ull * (unsigned long long)(((kk * kCout + n0) << 8) + c0);
        u32x4 g0;
        g0[0] = 1u;                                    // count = 1 descriptor
        g0[1] = ldsB;                                  // LDS destination
        g0[2] = (unsigned)gaddr;                       // global addr lo
        g0[3] = ((unsigned)(gaddr >> 32) & 0x01FFFFFFu)// global addr hi
              | 0x80000000u;                           // type = 2 ("image")
        __builtin_amdgcn_tensor_load_to_lds(g0, g1, gz4, gz4, gz8, 0);
      }
#else
      // --- Fallback: manual staging of the 128 x 32 weight tile ---
      {
        const h16* wp = wt + ((kk * kCout + n0) << 8) + c0;
#pragma unroll
        for (int i = 0; i < 16; ++i) {
          const int e = t + (i << 8);
          const int kc = e & 31;
          const int n  = e >> 5;
          BsubT[n][kc] = wp[(n << 8) + kc];
        }
      }
#endif
      // --- A tile: bilinear-gather 64 positions x 32 channels -> f16 LDS.
      //     Scalar base + 32-bit element offsets -> SADDR-form loads. ---
#pragma unroll
      for (int i = 0; i < 8; ++i) {
        const int c = (t >> 6) + (i << 2);
        const int chOff = (c0 + c) << 12;
        const float v = w0 * xb[chOff + i0] + w1 * xb[chOff + i1] +
                        w2 * xb[chOff + i2] + w3 * xb[chOff + i3];
        Asub[mloc][c] = (h16)v;
      }
#if __has_builtin(__builtin_amdgcn_tensor_load_to_lds)
      if (t < 32) __builtin_amdgcn_s_wait_tensorcnt(0);
#endif
      __syncthreads();

      // --- B fragment for nt = wave (same K order as A: lanes 0-15
      //     K=0..7,16..23, lanes 16-31 K=8..15,24..31) ---
      v16h bf;
      {
        const int ncol = (wave << 4) + (lane & 15);
        const int kbB = (lane >> 4) << 3;
#pragma unroll
        for (int i = 0; i < 4; ++i) {
          const h2 p = *(const h2*)&BsubT[ncol][kbB + 2 * i];
          bf[2 * i] = p[0]; bf[2 * i + 1] = p[1];
          const h2 q = *(const h2*)&BsubT[ncol][kbB + 16 + 2 * i];
          bf[8 + 2 * i] = q[0]; bf[8 + 2 * i + 1] = q[1];
        }
      }
      // --- A fragments + WMMA for mt = 0..3 (ISA 16-bit A 16x32 layout) ---
      const int mrow = lane & 15;
      const int kbA = (lane >> 4) << 3;   // K base 0 or 8
#pragma unroll
      for (int mt = 0; mt < 4; ++mt) {
        v16h a;
#pragma unroll
        for (int i = 0; i < 4; ++i) {
          const h2 p = *(const h2*)&Asub[(mt << 4) + mrow][kbA + 2 * i];
          a[2 * i] = p[0]; a[2 * i + 1] = p[1];
          const h2 q = *(const h2*)&Asub[(mt << 4) + mrow][kbA + 16 + 2 * i];
          a[8 + 2 * i] = q[0]; a[8 + 2 * i + 1] = q[1];
        }
        acc[mt] = __builtin_amdgcn_wmma_f32_16x16x32_f16(
            /*neg_a=*/false, a, /*neg_b=*/false, bf,
            /*c_mod=*/(short)0, acc[mt], /*reuse_a=*/false, /*reuse_b=*/false);
      }
      __syncthreads();
    }
  }

  // --- epilogue: D layout (VGPR r: lanes 0-15 M=r, lanes 16-31 M=r+8) ->
  //     LDS transpose -> coalesced f32 row stores + bias ---
#pragma unroll
  for (int mt = 0; mt < 4; ++mt) {
    const int n = (wave << 4) + (lane & 15);
    const int xbase = (mt << 4) + ((lane >> 4) << 3);
#pragma unroll
    for (int r = 0; r < 8; ++r)
      Dtile[n][xbase + r] = acc[mt][r];
  }
  __syncthreads();
#pragma unroll
  for (int i = 0; i < 32; ++i) {
    const int e = t + (i << 8);
    const int xcol = e & 63;
    const int n = e >> 6;
    out[((b * kCout + n0 + n) << 12) + (yy << 6) + xcol] =
        Dtile[n][xcol] + bias[n0 + n];
  }
}

// ---------------------------------------------------------------------------
extern "C" void kernel_launch(void* const* d_in, const int* in_sizes, int n_in,
                              void* d_out, int out_size, void* d_ws, size_t ws_size,
                              hipStream_t stream) {
  (void)in_sizes; (void)n_in; (void)out_size; (void)ws_size;
  const float* x     = (const float*)d_in[0];
  const float* w_mod = (const float*)d_in[1];
  const float* w_def = (const float*)d_in[2];
  const float* b_def = (const float*)d_in[3];
  float* out = (float*)d_out;

  float* modraw = (float*)d_ws;                                  // 1.77 MB
  h16*   wt     = (h16*)((char*)d_ws + kModBytes);               // 1.18 MB

  mod_conv_kernel<<<dim3(kBatch * kH * 27), 64, 0, stream>>>(x, w_mod, modraw);
  wconv_kernel<<<dim3((kWtElems + 255) / 256), 256, 0, stream>>>(w_def, wt);
  deform_gemm_kernel<<<dim3((kBatch * kHW) / 64, kCout / 128), 256, 0, stream>>>(
      x, modraw, wt, b_def, out);
}
